// STSGCL_81810537054352
// MI455X (gfx1250) — matrix operand order repaired and checked by reading, not compiled, not run on hardware
//
#include <hip/hip_runtime.h>
#include <hip/hip_bf16.h>
#include <math.h>
#include <float.h>

#define NN   2000
#define TT   12
#define BB   4
#define CC   32
#define WW   10      // W_WIN = T - STRIDES + 1
#define LL   3
#define M3   6000    // 3*N
#define KP   6016    // K (and padded M) rounded to multiple of 32; 6016 = 94*64
#define JJ   128     // B*C columns

typedef __hip_bfloat16 bf16;
typedef __attribute__((ext_vector_type(16))) __bf16 v16bf;
typedef __attribute__((ext_vector_type(8)))  float  v8f;

union FragBF { v16bf v; uint4 u[2]; };

__device__ __forceinline__ float sigf(float x) { return 1.0f / (1.0f + __expf(-x)); }

// ---------------------------------------------------------------------------
// Init: mid-max accumulator to -inf, zero the K-padding columns of both
// transposed-h ping-pong buffers (cols n = 6000..6015 are never written by
// the layer kernels but are read as K inputs of the next layer).
// ---------------------------------------------------------------------------
__global__ void init_kernel(float* __restrict__ midmax,
                            bf16* __restrict__ hT0, bf16* __restrict__ hT1) {
    size_t id = (size_t)blockIdx.x * 256 + threadIdx.x;
    const size_t nmid = (size_t)BB * WW * NN * CC;     // 2,560,000
    const size_t npad = (size_t)WW * JJ * (KP - M3);   // 20,480 per buffer
    if (id < nmid) { midmax[id] = -FLT_MAX; return; }
    id -= nmid;
    if (id < 2 * npad) {
        bf16* buf = (id < npad) ? hT0 : hT1;
        size_t p  = id % npad;
        size_t wj = p / (KP - M3);       // w*128 + j
        size_t pc = p % (KP - M3);
        buf[wj * KP + M3 + pc] = __float2bfloat16(0.0f);
    }
}

// ---------------------------------------------------------------------------
// Build padded bf16 adjacency (6016 x 6016, row-major, zero-padded) with the
// four temporal_adj block overwrites.
// ---------------------------------------------------------------------------
__global__ void build_A_kernel(const float* __restrict__ adj,
                               const float* __restrict__ tadj,
                               bf16* __restrict__ A) {
    size_t id = (size_t)blockIdx.x * 256 + threadIdx.x;
    if (id >= (size_t)KP * KP) return;
    int n = (int)(id / KP);
    int m = (int)(id % KP);
    float v = 0.0f;
    if (n < M3 && m < M3) {
        const bool rlo = n < NN,     rhi = n >= 2 * NN;
        const bool clo = m < NN,     chi = m >= 2 * NN;
        if      (rlo && clo) v = tadj[(size_t)n * NN + m];
        else if (rhi && chi) v = tadj[(size_t)(n - 2 * NN) * NN + (m - 2 * NN)];
        else if (rhi && clo) v = tadj[(size_t)(n - 2 * NN) * NN + m];
        else if (rlo && chi) v = tadj[(size_t)n * NN + (m - 2 * NN)];
        else                 v = adj [(size_t)n * M3 + m];
    }
    A[id] = __float2bfloat16(v);
}

// ---------------------------------------------------------------------------
// Build h0 transposed: hT0[(w*128 + b*32 + c) * 6016 + (s*2000 + node)]
//   = x[b, w+s, node, c] + temporal_emb[w+s, c] + spatial_emb[node, c]
// ---------------------------------------------------------------------------
__global__ void build_h0_kernel(const float* __restrict__ x,
                                const float* __restrict__ temb,
                                const float* __restrict__ semb,
                                bf16* __restrict__ hT0) {
    size_t id = (size_t)blockIdx.x * 256 + threadIdx.x;
    const size_t total = (size_t)WW * JJ * KP;
    if (id >= total) return;
    int w   = (int)(id / ((size_t)JJ * KP));
    int rem = (int)(id % ((size_t)JJ * KP));
    int j   = rem / KP;
    int m   = rem % KP;
    float v = 0.0f;
    if (m < M3) {
        int s = m / NN, node = m % NN;
        int t = w + s;
        int b = j >> 5, c = j & 31;
        v = x[(((size_t)b * TT + t) * NN + node) * CC + c]
          + temb[(size_t)t * CC + c]
          + semb[(size_t)node * CC + c];
    }
    hT0[id] = __float2bfloat16(v);
}

// ---------------------------------------------------------------------------
// Fused per-layer kernel.  Block = 256 threads (8 waves) computes a
// 64-row x 128-col strip of D = A(6016x6016,bf16) @ h_in(6016x128,bf16).
// Each wave register-blocks a 32x32 tile: 2 A-frags + 2 B-frags (8 b128
// loads) feed 4 v_wmma_f32_16x16x32_bf16 per K-step -> 2 loads/WMMA.
// Then FC(32->64) + bias + GLU from LDS, write transposed bf16 h_out and
// fold the middle block into a running max in final (B,W,N,C) layout.
// Grid = (6016/64 = 94, 10 windows).
// ---------------------------------------------------------------------------
__global__ __launch_bounds__(256) void layer_kernel(
        const bf16*  __restrict__ A,
        const bf16*  __restrict__ hin,
        const float* __restrict__ Wfc,
        const float* __restrict__ bfc,
        bf16*        __restrict__ hout,
        float*       __restrict__ midmax,
        int layer) {
    __shared__ float g[64][JJ + 4];      // 64 x 128 f32 result strip (~34 KB)
    __shared__ float sW[CC * 2 * CC];    // 32 x 64 FC weights for (win, layer)
    __shared__ float sb[2 * CC];

    const int win     = blockIdx.y;
    const int rowbase = blockIdx.x * 64;
    const int tid     = threadIdx.x;

    const float* Wsrc = Wfc + (size_t)(win * LL + layer) * (CC * 2 * CC);
    for (int i = tid; i < CC * 2 * CC; i += 256) sW[i] = Wsrc[i];
    if (tid < 2 * CC) sb[tid] = bfc[(size_t)(win * LL + layer) * (2 * CC) + tid];

    const int lane = tid & 31;
    const int wave = tid >> 5;
    const int half = lane >> 4;      // K-half select per ISA 16-bit layouts
    const int l16  = lane & 15;
    const int rowg = wave >> 2;      // 0..1  : which 32-row group
    const int colg = wave & 3;       // 0..3  : which 32-col group

    // A-fragments: lane = M row (l16); K chunks {8h..8h+7} and {16+8h..+7}
    const bf16* Arow0 = A + (size_t)(rowbase + 32 * rowg + l16) * KP;
    const bf16* Arow1 = Arow0 + (size_t)16 * KP;
    // B-fragments: lane = N col (l16) of transposed h; K chunks {8h..8h+7} and {16+8h..+7}
    const bf16* Brow0 = hin + (size_t)(win * JJ + 32 * colg + l16) * KP;
    const bf16* Brow1 = Brow0 + (size_t)16 * KP;

    v8f acc00 = {0.f,0.f,0.f,0.f,0.f,0.f,0.f,0.f};
    v8f acc01 = acc00, acc10 = acc00, acc11 = acc00;

    for (int kk = 0; kk < KP; kk += 32) {
        FragBF a0, a1, b0, b1;
        a0.u[0] = *(const uint4*)(Arow0 + kk + 8 * half);
        a0.u[1] = *(const uint4*)(Arow0 + kk + 16 + 8 * half);
        a1.u[0] = *(const uint4*)(Arow1 + kk + 8 * half);
        a1.u[1] = *(const uint4*)(Arow1 + kk + 16 + 8 * half);
        b0.u[0] = *(const uint4*)(Brow0 + kk + 8 * half);
        b0.u[1] = *(const uint4*)(Brow0 + kk + 16 + 8 * half);
        b1.u[0] = *(const uint4*)(Brow1 + kk + 8 * half);
        b1.u[1] = *(const uint4*)(Brow1 + kk + 16 + 8 * half);
        acc00 = __builtin_amdgcn_wmma_f32_16x16x32_bf16(
                    false, a0.v, false, b0.v, (short)0, acc00, false, false);
        acc01 = __builtin_amdgcn_wmma_f32_16x16x32_bf16(
                    false, a0.v, false, b1.v, (short)0, acc01, false, false);
        acc10 = __builtin_amdgcn_wmma_f32_16x16x32_bf16(
                    false, a1.v, false, b0.v, (short)0, acc10, false, false);
        acc11 = __builtin_amdgcn_wmma_f32_16x16x32_bf16(
                    false, a1.v, false, b1.v, (short)0, acc11, false, false);
    }

    // C/D layout: lanes 0-15 -> M = r, lanes 16-31 -> M = r + 8; N = l16
    {
        const int rb = 32 * rowg;
        const int cb = 32 * colg;
        #pragma unroll
        for (int r = 0; r < 8; ++r) {
            const int mr = r + 8 * half;
            g[rb + mr     ][cb + l16     ] = acc00[r];
            g[rb + mr     ][cb + 16 + l16] = acc01[r];
            g[rb + 16 + mr][cb + l16     ] = acc10[r];
            g[rb + 16 + mr][cb + 16 + l16] = acc11[r];
        }
    }
    __syncthreads();

    // FC + GLU over the 64 x (4 x 32) strip: 8192 outputs, 32 per thread.
    for (int o = tid; o < 64 * JJ; o += 256) {
        const int nl = o >> 7;
        const int n  = rowbase + nl;
        if (n >= M3) continue;           // padded rows 6000..6015
        const int bc = o & (JJ - 1);
        const int b  = bc >> 5, c = bc & 31;
        const float* grow = &g[nl][b * CC];
        float lhs = sb[c], rhs = sb[c + CC];
        #pragma unroll
        for (int cin = 0; cin < CC; ++cin) {
            const float hv = grow[cin];
            lhs += hv * sW[cin * 2 * CC + c];
            rhs += hv * sW[cin * 2 * CC + c + CC];
        }
        const float val = lhs * sigf(rhs);
        hout[(size_t)(win * JJ + bc) * KP + n] = __float2bfloat16(val);
        if (n >= NN && n < 2 * NN) {   // middle block -> layer max, (B,W,N,C)
            const size_t mi = (((size_t)b * WW + win) * NN + (n - NN)) * CC + c;
            const float old = midmax[mi];
            midmax[mi] = val > old ? val : old;
        }
    }
}

// ---------------------------------------------------------------------------
// GRU scan: 8000 independent cells (b,n), each thread runs all 10 steps with
// its 32-wide hidden state in registers; weights staged in LDS.
// Writes d_out = out + residual directly.
// ---------------------------------------------------------------------------
__global__ __launch_bounds__(256) void gru_kernel(
        const float* __restrict__ out,
        const float* __restrict__ Wr, const float* __restrict__ br,
        const float* __restrict__ Wu, const float* __restrict__ bu,
        const float* __restrict__ Wn, const float* __restrict__ bn,
        const float* __restrict__ Wo, const float* __restrict__ bo,
        float* __restrict__ y) {
    __shared__ float sWr[64 * 32], sWu[64 * 32], sWn[64 * 32], sWo[32 * 32];
    __shared__ float sbr[32], sbu[32], sbn[32], sbo[32];
    const int tid = threadIdx.x;
    for (int i = tid; i < 64 * 32; i += 256) {
        sWr[i] = Wr[i]; sWu[i] = Wu[i]; sWn[i] = Wn[i];
    }
    for (int i = tid; i < 32 * 32; i += 256) sWo[i] = Wo[i];
    if (tid < 32) { sbr[tid] = br[tid]; sbu[tid] = bu[tid];
                    sbn[tid] = bn[tid]; sbo[tid] = bo[tid]; }
    __syncthreads();

    const int cell = blockIdx.x * 256 + tid;
    if (cell >= BB * NN) return;
    const int b = cell / NN, n = cell % NN;

    float hid[32];
    #pragma unroll
    for (int c = 0; c < 32; ++c) hid[c] = 0.0f;

    for (int t = 0; t < WW; ++t) {
        const float* inp = out + (((size_t)b * WW + t) * NN + n) * CC;
        float xi[32];
        for (int c = 0; c < 32; ++c) xi[c] = inp[c];

        float rst[32], upd[32];
        for (int c = 0; c < 32; ++c) {
            float r = sbr[c], u = sbu[c];
            for (int k = 0; k < 32; ++k) {
                r += xi[k] * sWr[k * 32 + c] + hid[k] * sWr[(32 + k) * 32 + c];
                u += xi[k] * sWu[k * 32 + c] + hid[k] * sWu[(32 + k) * 32 + c];
            }
            rst[c] = sigf(r); upd[c] = sigf(u);
        }
        float nm[32];
        for (int c = 0; c < 32; ++c) {
            float v = sbn[c];
            for (int k = 0; k < 32; ++k)
                v += xi[k] * sWn[k * 32 + c]
                   + (rst[k] * hid[k]) * sWn[(32 + k) * 32 + c];
            nm[c] = tanhf(v);
        }
        float outp[32];
        for (int c = 0; c < 32; ++c)
            outp[c] = upd[c] * hid[c] + (1.0f - upd[c]) * nm[c];

        float* yp = y + (((size_t)b * WW + t) * NN + n) * CC;
        for (int c = 0; c < 32; ++c) {
            float yv = sbo[c];
            for (int k = 0; k < 32; ++k) yv += outp[k] * sWo[k * 32 + c];
            yp[c] = xi[c] + sigf(yv);
        }
        for (int c = 0; c < 32; ++c) hid[c] = outp[c];
    }
}

// ---------------------------------------------------------------------------
extern "C" void kernel_launch(void* const* d_in, const int* in_sizes, int n_in,
                              void* d_out, int out_size, void* d_ws, size_t ws_size,
                              hipStream_t stream) {
    (void)in_sizes; (void)n_in; (void)out_size; (void)ws_size;
    const float* x    = (const float*)d_in[0];
    const float* adj  = (const float*)d_in[1];
    const float* tadj = (const float*)d_in[2];
    const float* temb = (const float*)d_in[3];
    const float* semb = (const float*)d_in[4];
    const float* Wfc  = (const float*)d_in[5];
    const float* bfc  = (const float*)d_in[6];
    const float* Wr   = (const float*)d_in[7];
    const float* brr  = (const float*)d_in[8];
    const float* Wu   = (const float*)d_in[9];
    const float* bu   = (const float*)d_in[10];
    const float* Wn   = (const float*)d_in[11];
    const float* bn   = (const float*)d_in[12];
    const float* Wo   = (const float*)d_in[13];
    const float* bo   = (const float*)d_in[14];
    float* y = (float*)d_out;

    char* ws = (char*)d_ws;
    size_t off = 0;
    auto alloc = [&](size_t bytes) -> char* {
        char* p = ws + off;
        off = (off + bytes + 255) & ~(size_t)255;
        return p;
    };
    bf16*  Abf    = (bf16*) alloc((size_t)KP * KP * sizeof(bf16));      // 72.4 MB
    bf16*  hT0    = (bf16*) alloc((size_t)WW * JJ * KP * sizeof(bf16)); // 15.4 MB
    bf16*  hT1    = (bf16*) alloc((size_t)WW * JJ * KP * sizeof(bf16)); // 15.4 MB
    float* midmax = (float*)alloc((size_t)BB * WW * NN * CC * sizeof(float)); // 10.2 MB

    {
        size_t total = (size_t)BB * WW * NN * CC + 2 * (size_t)WW * JJ * (KP - M3);
        init_kernel<<<(unsigned)((total + 255) / 256), 256, 0, stream>>>(midmax, hT0, hT1);
    }
    {
        size_t total = (size_t)KP * KP;
        build_A_kernel<<<(unsigned)((total + 255) / 256), 256, 0, stream>>>(adj, tadj, Abf);
    }
    {
        size_t total = (size_t)WW * JJ * KP;
        build_h0_kernel<<<(unsigned)((total + 255) / 256), 256, 0, stream>>>(x, temb, semb, hT0);
    }

    dim3 grid(KP / 64, WW);   // 94 x 10
    layer_kernel<<<grid, 256, 0, stream>>>(Abf, hT0, Wfc, bfc, hT1, midmax, 0);
    layer_kernel<<<grid, 256, 0, stream>>>(Abf, hT1, Wfc, bfc, hT0, midmax, 1);
    layer_kernel<<<grid, 256, 0, stream>>>(Abf, hT0, Wfc, bfc, hT1, midmax, 2);

    gru_kernel<<<(BB * NN + 255) / 256, 256, 0, stream>>>(
        midmax, Wr, brr, Wu, bu, Wn, bn, Wo, bo, y);
}
